// LinearTransformer_32899449487832
// MI455X (gfx1250) — matrix operand order repaired and checked by reading, not compiled, not run on hardware
//
#include <hip/hip_runtime.h>
#include <hip/hip_bf16.h>
#include <math.h>

// Problem constants (match reference)
#define BDIM   2
#define CCH    512
#define NSP    4096      // H*W
#define HEADS  8
#define DHEAD  64
#define HIDN   512
#define GROUPS 32
#define CPG    16        // channels per group
#define EPSV   1e-5f
#define QK_SCALE 0.35355339059327373f  // 64^-0.25, applied to BOTH q and k

typedef __bf16 bf16;
typedef __attribute__((ext_vector_type(16))) __bf16 v16bf;
typedef __attribute__((ext_vector_type(8)))  __bf16 v8bf;
typedef __attribute__((ext_vector_type(8)))  float  v8f;

static __device__ __forceinline__ v8f zero_v8f() {
  v8f z;
#pragma unroll
  for (int i = 0; i < 8; ++i) z[i] = 0.0f;
  return z;
}

static __device__ __forceinline__ v8f wmma_bf16(v16bf a, v16bf b, v8f c) {
  // (neg_a, A, neg_b, B, c_mod, C, reuse_a, reuse_b)
  return __builtin_amdgcn_wmma_f32_16x16x32_bf16(false, a, false, b, (short)0, c,
                                                 false, false);
}

// gfx1250 async global->LDS copy, 16B per lane, tracked by ASYNCcnt.
static __device__ __forceinline__ void async_b128(unsigned lds_off, const void* g) {
  asm volatile("global_load_async_to_lds_b128 %0, %1, off"
               :: "v"(lds_off), "v"((unsigned long long)(uintptr_t)g)
               : "memory");
}
static __device__ __forceinline__ void wait_async0() {
  asm volatile("s_wait_asynccnt 0x0" ::: "memory");
}

// ---- WMMA fragment loaders (contiguous-k order, same for A and B, wave32) ----------
// A (16x32, bf16, row-major source [row][ld]):
//   lane L: M = m0 + (L&15); elems e -> K = k0 + (L>>4)*16 + e.
//   One contiguous 32B read per lane.
static __device__ __forceinline__ v16bf load_A(const bf16* base, int ld, int m0,
                                               int k0, int lane) {
  const bf16* row = base + (size_t)(m0 + (lane & 15)) * ld + k0 + ((lane >> 4) << 4);
  return *(const v16bf*)(row);
}

// B (32x16 KxN) loaded from B-transposed storage (row-major [N][K], ld = K stride):
//   lane L: N = n0 + (L&15); elems e -> K = k0 + (L>>4)*16 + e.
//   One contiguous 32B read per lane.
static __device__ __forceinline__ v16bf load_Bt(const bf16* baseT, int ld, int k0,
                                                int n0, int lane) {
  const bf16* row = baseT + (size_t)(n0 + (lane & 15)) * ld + k0 + ((lane >> 4) << 4);
  return *(const v16bf*)(row);
}
// C/D (16x16 f32): lane L elem e -> M = e + (L>>4)*8, N = L&15.

// ---------------------------------------------------------------------------
// K0: convert fp32 weights -> bf16 workspace copies
__global__ __launch_bounds__(256) void cvt_w_kernel(
    const float* wq, const float* wk, const float* wv, const float* wo,
    bf16* wqb, bf16* wkb, bf16* wvb, bf16* wob) {
  int i = blockIdx.x * 256 + threadIdx.x;
  if (i < CCH * HIDN) {
    wqb[i] = (bf16)wq[i];
    wkb[i] = (bf16)wk[i];
    wvb[i] = (bf16)wv[i];
    wob[i] = (bf16)wo[i];
  }
}

// ---------------------------------------------------------------------------
// K1: GroupNorm, writing x-normed TRANSPOSED as bf16: xnT[b][n][c].
// Pass 2: each thread owns spatial positions and emits one contiguous 32B row
// of the group's 16 channels per position.
__global__ __launch_bounds__(256) void gn_kernel(
    const float* __restrict__ x, const float* __restrict__ gamma,
    const float* __restrict__ beta, bf16* __restrict__ xnT) {
  const int g = blockIdx.x, b = blockIdx.y, tid = threadIdx.x;
  const float* xg = x + ((size_t)b * CCH + (size_t)g * CPG) * NSP;
  float s = 0.f, s2 = 0.f;
  for (int i = tid; i < CPG * NSP; i += 256) {
    float v = xg[i];
    s += v; s2 += v * v;
  }
  __shared__ float red[256], red2[256];
  red[tid] = s; red2[tid] = s2;
  __syncthreads();
  for (int off = 128; off > 0; off >>= 1) {
    if (tid < off) { red[tid] += red[tid + off]; red2[tid] += red2[tid + off]; }
    __syncthreads();
  }
  const float inv_cnt = 1.0f / (float)(CPG * NSP);
  const float mu = red[0] * inv_cnt;
  const float var = red2[0] * inv_cnt - mu * mu;
  const float rstd = rsqrtf(var + EPSV);

  float gm[CPG], bt[CPG];
#pragma unroll
  for (int c = 0; c < CPG; ++c) {
    gm[c] = gamma[g * CPG + c] * rstd;
    bt[c] = beta[g * CPG + c];
  }
  for (int n = tid; n < NSP; n += 256) {
    v16bf row;
#pragma unroll
    for (int c = 0; c < CPG; ++c)
      row[c] = (bf16)((xg[(size_t)c * NSP + n] - mu) * gm[c] + bt[c]);
    *(v16bf*)(xnT + ((size_t)b * NSP + n) * CCH + g * CPG) = row;
  }
}

// ---------------------------------------------------------------------------
// K2: fused QKV projection. Wave owns a 16(m) x 64(n) strip; the xnT
// B-fragment is loaded once per (k-step, n-tile) and reused by Q/K/V WMMAs
// (12 WMMA per 14 b128 loads).  Q,K -> [b][h][n][d] bf16 (pre-scaled);
// V -> [b][h][d][n] bf16.
__global__ __launch_bounds__(128) void qkv_kernel(
    const bf16* __restrict__ xnT, const bf16* __restrict__ wqb,
    const bf16* __restrict__ wkb, const bf16* __restrict__ wvb,
    const float* __restrict__ bq, const float* __restrict__ bk,
    const float* __restrict__ bv, bf16* __restrict__ qO,
    bf16* __restrict__ kO, bf16* __restrict__ vO) {
  const int lane = threadIdx.x & 31, wave = threadIdx.x >> 5;
  const int n0 = blockIdx.x * 64;
  const int m0 = blockIdx.y * 64 + wave * 16;
  const int b = blockIdx.z;
  const bf16* Bb = xnT + (size_t)b * NSP * CCH;

  v8f accq[4], acck[4], accv[4];
#pragma unroll
  for (int t = 0; t < 4; ++t) { accq[t] = zero_v8f(); acck[t] = zero_v8f(); accv[t] = zero_v8f(); }

#pragma unroll 2
  for (int k0 = 0; k0 < CCH; k0 += 32) {
    v16bf b0 = load_Bt(Bb, CCH, k0, n0 +  0, lane);
    v16bf b1 = load_Bt(Bb, CCH, k0, n0 + 16, lane);
    v16bf b2 = load_Bt(Bb, CCH, k0, n0 + 32, lane);
    v16bf b3 = load_Bt(Bb, CCH, k0, n0 + 48, lane);
    v16bf aq = load_A(wqb, CCH, m0, k0, lane);
    v16bf ak = load_A(wkb, CCH, m0, k0, lane);
    v16bf av = load_A(wvb, CCH, m0, k0, lane);
    accq[0] = wmma_bf16(aq, b0, accq[0]);
    accq[1] = wmma_bf16(aq, b1, accq[1]);
    accq[2] = wmma_bf16(aq, b2, accq[2]);
    accq[3] = wmma_bf16(aq, b3, accq[3]);
    acck[0] = wmma_bf16(ak, b0, acck[0]);
    acck[1] = wmma_bf16(ak, b1, acck[1]);
    acck[2] = wmma_bf16(ak, b2, acck[2]);
    acck[3] = wmma_bf16(ak, b3, acck[3]);
    accv[0] = wmma_bf16(av, b0, accv[0]);
    accv[1] = wmma_bf16(av, b1, accv[1]);
    accv[2] = wmma_bf16(av, b2, accv[2]);
    accv[3] = wmma_bf16(av, b3, accv[3]);
  }

  const int h = m0 >> 6;                       // tile never crosses a head
  const int d0 = (m0 & (DHEAD - 1)) + ((lane >> 4) << 3);
  float bqm[8], bkm[8], bvm[8];
#pragma unroll
  for (int e = 0; e < 8; ++e) {
    int m = m0 + e + ((lane >> 4) << 3);
    bqm[e] = bq[m]; bkm[e] = bk[m]; bvm[e] = bv[m];
  }
#pragma unroll
  for (int nt = 0; nt < 4; ++nt) {
    const int ncol = n0 + nt * 16 + (lane & 15);
    v8bf qrow, krow;
#pragma unroll
    for (int e = 0; e < 8; ++e) {
      qrow[e] = (bf16)((accq[nt][e] + bqm[e]) * QK_SCALE);
      krow[e] = (bf16)((acck[nt][e] + bkm[e]) * QK_SCALE);
    }
    size_t qk_idx = (((size_t)b * HEADS + h) * NSP + ncol) * DHEAD + d0;
    *(v8bf*)(qO + qk_idx) = qrow;              // contiguous 16B (d-major)
    *(v8bf*)(kO + qk_idx) = krow;
#pragma unroll
    for (int e = 0; e < 8; ++e)
      vO[(((size_t)b * HEADS + h) * DHEAD + d0 + e) * NSP + ncol] =
          (bf16)(accv[nt][e] + bvm[e]);
  }
}

// ---------------------------------------------------------------------------
// K3: flash attention per (b,h). 128 threads = 4 waves; each wave owns 16
// queries (full d=64). K/V 64x64 tiles are DOUBLE-BUFFERED in LDS and filled
// with gfx1250 async global->LDS copies (ASYNCcnt), overlapping the WMMA work
// on the current tile with the DMA of the next.
__global__ __launch_bounds__(128) void attn_kernel(
    const bf16* __restrict__ q, const bf16* __restrict__ k,
    const bf16* __restrict__ v, bf16* __restrict__ aoT) {
  __shared__ __attribute__((aligned(32))) bf16 Ksh[2][64 * 64];    // [key][d]
  __shared__ __attribute__((aligned(32))) bf16 Vsh[2][64 * 64];    // [d][key]
  __shared__ __attribute__((aligned(32))) bf16 Psh[4][16 * 64];    // per-wave P

  const int tid = threadIdx.x, lane = tid & 31, wave = tid >> 5;
  const int qt = blockIdx.x, h = blockIdx.y, b = blockIdx.z;
  const bf16* qb = q + ((size_t)b * HEADS + h) * NSP * DHEAD;
  const bf16* kb = k + ((size_t)b * HEADS + h) * NSP * DHEAD;
  const bf16* vb = v + ((size_t)b * HEADS + h) * DHEAD * NSP;
  const int q0 = qt * 64 + wave * 16;

  // Q fragments persist across the whole key loop.
  const v16bf qf0 = load_A(qb, DHEAD, q0, 0, lane);
  const v16bf qf1 = load_A(qb, DHEAD, q0, 32, lane);

  float rmaxv[8], rsumv[8];
  v8f acc[4];
#pragma unroll
  for (int e = 0; e < 8; ++e) { rmaxv[e] = -3.0e38f; rsumv[e] = 0.0f; }
#pragma unroll
  for (int t = 0; t < 4; ++t) acc[t] = zero_v8f();

  // Async-stage one 64-key tile pair (K: contiguous 8KB; V: 64 rows x 128B).
  auto stage_tiles = [&](int buf, int kt0) {
    unsigned kbase = (unsigned)(uintptr_t)&Ksh[buf][0];
    unsigned vbase = (unsigned)(uintptr_t)&Vsh[buf][0];
    const bf16* ksrc = kb + (size_t)kt0 * DHEAD;
#pragma unroll
    for (int j = 0; j < 4; ++j) {
      int i = tid + 128 * j;
      async_b128(kbase + i * 16, ksrc + i * 8);
      int r = i >> 3, c8 = i & 7;
      async_b128(vbase + i * 16, vb + (size_t)r * NSP + kt0 + c8 * 8);
    }
  };

  stage_tiles(0, 0);
  wait_async0();
  __syncthreads();

  for (int kt = 0; kt < NSP / 64; ++kt) {
    const int buf = kt & 1;
    if (kt + 1 < NSP / 64) stage_tiles(buf ^ 1, (kt + 1) * 64);  // prefetch

    // S = Q Kt : 4 tiles of 16x16, each 2 chained WMMAs over d.
    v8f s[4];
#pragma unroll
    for (int nt = 0; nt < 4; ++nt) {
      v8f sv = zero_v8f();
      sv = wmma_bf16(qf0, load_Bt(Ksh[buf], 64, 0, nt * 16, lane), sv);
      sv = wmma_bf16(qf1, load_Bt(Ksh[buf], 64, 32, nt * 16, lane), sv);
      s[nt] = sv;
    }

    // Online softmax. Row m = e + (lane>>4)*8 is spread across 16 lanes.
    bf16* pw_base = &Psh[wave][0];
#pragma unroll
    for (int e = 0; e < 8; ++e) {
      float m4 = fmaxf(fmaxf(s[0][e], s[1][e]), fmaxf(s[2][e], s[3][e]));
#pragma unroll
      for (int msk = 8; msk; msk >>= 1) m4 = fmaxf(m4, __shfl_xor(m4, msk, 32));
      float mnew = fmaxf(rmaxv[e], m4);
      float corr = __expf(rmaxv[e] - mnew);
      rmaxv[e] = mnew;
      float p0 = __expf(s[0][e] - mnew);
      float p1 = __expf(s[1][e] - mnew);
      float p2 = __expf(s[2][e] - mnew);
      float p3 = __expf(s[3][e] - mnew);
      float ps = p0 + p1 + p2 + p3;
#pragma unroll
      for (int msk = 8; msk; msk >>= 1) ps += __shfl_xor(ps, msk, 32);
      rsumv[e] = rsumv[e] * corr + ps;
      acc[0][e] *= corr; acc[1][e] *= corr; acc[2][e] *= corr; acc[3][e] *= corr;
      // Transpose-through-LDS: C-layout -> row-major [16][64] for A-fragments.
      bf16* pw = pw_base + (e + ((lane >> 4) << 3)) * 64 + (lane & 15);
      pw[0] = (bf16)p0; pw[16] = (bf16)p1; pw[32] = (bf16)p2; pw[48] = (bf16)p3;
    }

    // O += P V : 4 d-tiles, 2 chained WMMAs over keys each. Per-wave LDS; DS
    // ops from one wave complete in order, so no cross-wave barrier needed.
#pragma unroll
    for (int dt = 0; dt < 4; ++dt) {
      acc[dt] = wmma_bf16(load_A(pw_base, 64, 0, 0, lane),
                          load_Bt(Vsh[buf], 64, 0, dt * 16, lane), acc[dt]);
      acc[dt] = wmma_bf16(load_A(pw_base, 64, 0, 32, lane),
                          load_Bt(Vsh[buf], 64, 32, dt * 16, lane), acc[dt]);
    }

    wait_async0();     // our prefetch for buf^1 has landed in LDS
    __syncthreads();   // everyone done reading buf before it is overwritten
  }

  // Normalize + store as aoT[b][n][hid] bf16 (Bt layout for out-projection).
#pragma unroll
  for (int e = 0; e < 8; ++e) {
    float inv = 1.0f / rsumv[e];
    int m = e + ((lane >> 4) << 3);
    int nrow = q0 + m;
    bf16* dst = aoT + ((size_t)b * NSP + nrow) * HIDN + h * DHEAD + (lane & 15);
    dst[0]  = (bf16)(acc[0][e] * inv);
    dst[16] = (bf16)(acc[1][e] * inv);
    dst[32] = (bf16)(acc[2][e] * inv);
    dst[48] = (bf16)(acc[3][e] * inv);
  }
}

// ---------------------------------------------------------------------------
// K4: output projection + bias + residual, fp32 epilogue. Wave owns 16x64.
__global__ __launch_bounds__(128) void out_kernel(
    const bf16* __restrict__ aoT, const bf16* __restrict__ wob,
    const float* __restrict__ bo, const float* __restrict__ x,
    float* __restrict__ out) {
  const int lane = threadIdx.x & 31, wave = threadIdx.x >> 5;
  const int n0 = blockIdx.x * 64;
  const int m0 = blockIdx.y * 64 + wave * 16;
  const int b = blockIdx.z;
  const bf16* Bb = aoT + (size_t)b * NSP * HIDN;

  v8f acc[4];
#pragma unroll
  for (int t = 0; t < 4; ++t) acc[t] = zero_v8f();

#pragma unroll 2
  for (int k0 = 0; k0 < HIDN; k0 += 32) {
    v16bf a = load_A(wob, HIDN, m0, k0, lane);
    acc[0] = wmma_bf16(a, load_Bt(Bb, HIDN, k0, n0 +  0, lane), acc[0]);
    acc[1] = wmma_bf16(a, load_Bt(Bb, HIDN, k0, n0 + 16, lane), acc[1]);
    acc[2] = wmma_bf16(a, load_Bt(Bb, HIDN, k0, n0 + 32, lane), acc[2]);
    acc[3] = wmma_bf16(a, load_Bt(Bb, HIDN, k0, n0 + 48, lane), acc[3]);
  }

  float bom[8];
#pragma unroll
  for (int e = 0; e < 8; ++e) bom[e] = bo[m0 + e + ((lane >> 4) << 3)];
#pragma unroll
  for (int nt = 0; nt < 4; ++nt) {
    const int ncol = n0 + nt * 16 + (lane & 15);
#pragma unroll
    for (int e = 0; e < 8; ++e) {
      int m = m0 + e + ((lane >> 4) << 3);
      size_t idx = ((size_t)b * CCH + m) * NSP + ncol;
      out[idx] = acc[nt][e] + bom[e] + x[idx];
    }
  }
}

// ---------------------------------------------------------------------------
extern "C" void kernel_launch(void* const* d_in, const int* in_sizes, int n_in,
                              void* d_out, int out_size, void* d_ws, size_t ws_size,
                              hipStream_t stream) {
  const float* x     = (const float*)d_in[0];
  const float* gamma = (const float*)d_in[1];
  const float* beta  = (const float*)d_in[2];
  const float* wq    = (const float*)d_in[3];
  const float* bq    = (const float*)d_in[4];
  const float* wk    = (const float*)d_in[5];
  const float* bk    = (const float*)d_in[6];
  const float* wv    = (const float*)d_in[7];
  const float* bv    = (const float*)d_in[8];
  const float* wo    = (const float*)d_in[9];
  const float* bo    = (const float*)d_in[10];
  float* out = (float*)d_out;

  // Workspace layout (bf16), all offsets 256B aligned.
  char* w = (char*)d_ws;
  const size_t SZ_ACT = (size_t)BDIM * NSP * CCH * sizeof(bf16);       // 8 MiB
  const size_t SZ_W   = (size_t)CCH * HIDN * sizeof(bf16);             // 0.5 MiB
  bf16* xnT = (bf16*)(w);                       w += SZ_ACT;
  bf16* qB  = (bf16*)(w);                       w += SZ_ACT;
  bf16* kB  = (bf16*)(w);                       w += SZ_ACT;
  bf16* vB  = (bf16*)(w);                       w += SZ_ACT;
  bf16* aoT = (bf16*)(w);                       w += SZ_ACT;
  bf16* wqb = (bf16*)(w);                       w += SZ_W;
  bf16* wkb = (bf16*)(w);                       w += SZ_W;
  bf16* wvb = (bf16*)(w);                       w += SZ_W;
  bf16* wob = (bf16*)(w);                       w += SZ_W;

  cvt_w_kernel<<<dim3((CCH * HIDN + 255) / 256), 256, 0, stream>>>(
      wq, wk, wv, wo, wqb, wkb, wvb, wob);

  gn_kernel<<<dim3(GROUPS, BDIM), 256, 0, stream>>>(x, gamma, beta, xnT);

  qkv_kernel<<<dim3(NSP / 64, HIDN / 64, BDIM), 128, 0, stream>>>(
      xnT, wqb, wkb, wvb, bq, bk, bv, qB, kB, vB);

  attn_kernel<<<dim3(NSP / 64, HEADS, BDIM), 128, 0, stream>>>(qB, kB, vB, aoT);

  out_kernel<<<dim3(NSP / 64, CCH / 64, BDIM), 128, 0, stream>>>(
      aoT, wob, bo, x, out);
}
